// Wassertein_Loss_15350213116490
// MI455X (gfx1250) — matrix operand order repaired and checked by reading, not compiled, not run on hardware
//
#include <hip/hip_runtime.h>
#include <hip/hip_bf16.h>

typedef __attribute__((ext_vector_type(16))) __bf16 v16bf;
typedef __attribute__((ext_vector_type(8)))  float  v8f;
typedef unsigned short u16;
typedef unsigned int   u32;

#define MAX_EPS 40
#define DIMS 32

union V16 { uint4 q[2]; v16bf v; };

__device__ inline u16 f2bf(float f) {
    u32 u = __float_as_uint(f);
    u32 r = (u + 0x7fffu + ((u >> 16) & 1u)) >> 16;
    return (u16)r;
}
__device__ inline float bf2f(u16 h) { return __uint_as_float(((u32)h) << 16); }

// ---------------------------------------------------------------------------
// Prep: bf16 hi/lo split of both point sets + squared norms
// ---------------------------------------------------------------------------
__global__ __launch_bounds__(256)
void prep_kernel(const float* __restrict__ X, const float* __restrict__ Y, int n,
                 u16* __restrict__ Xhi, u16* __restrict__ Xlo,
                 u16* __restrict__ Yhi, u16* __restrict__ Ylo,
                 float* __restrict__ x2, float* __restrict__ y2) {
    int i = blockIdx.x * blockDim.x + threadIdx.x;
    const float* src; u16* hi; u16* lo; float* sq; int row;
    if (i < n)          { src = X; hi = Xhi; lo = Xlo; sq = x2; row = i; }
    else if (i < 2 * n) { src = Y; hi = Yhi; lo = Ylo; sq = y2; row = i - n; }
    else return;
    float acc = 0.0f;
    #pragma unroll
    for (int d = 0; d < DIMS; ++d) {
        float v = src[row * DIMS + d];
        acc = fmaf(v, v, acc);
        u16 h = f2bf(v);
        float vh = bf2f(h);
        u16 l = f2bf(v - vh);
        hi[row * DIMS + d] = h;
        lo[row * DIMS + d] = l;
    }
    sq[row] = acc;
}

// ---------------------------------------------------------------------------
// Schedule: per-dim min/max over both sets -> diameter -> geomloss eps list
// ---------------------------------------------------------------------------
__global__ __launch_bounds__(1024)
void schedule_kernel(const float* __restrict__ X, const float* __restrict__ Y, int n,
                     float* __restrict__ epsBuf, int* __restrict__ kBuf) {
    __shared__ float smin[32][33];
    __shared__ float smax[32][33];
    int tid = threadIdx.x;
    int dim = tid & 31, grp = tid >> 5;
    float mn = __builtin_inff(), mx = -__builtin_inff();
    for (int row = grp; row < n; row += 32) {
        float v = X[row * DIMS + dim]; mn = fminf(mn, v); mx = fmaxf(mx, v);
        float w = Y[row * DIMS + dim]; mn = fminf(mn, w); mx = fmaxf(mx, w);
    }
    smin[grp][dim] = mn; smax[grp][dim] = mx;
    __syncthreads();
    if (tid < 32) {
        float a = __builtin_inff(), b = -__builtin_inff();
        for (int g = 0; g < 32; ++g) { a = fminf(a, smin[g][tid]); b = fmaxf(b, smax[g][tid]); }
        smin[0][tid] = b - a;  // per-dim range
    }
    __syncthreads();
    if (tid == 0) {
        double d2 = 0.0;
        for (int d = 0; d < DIMS; ++d) { double r = (double)smin[0][d]; d2 += r * r; }
        double diam = sqrt(d2);
        int k = 0;
        epsBuf[k++] = (float)(diam * diam);
        double start = 2.0 * log(diam);
        double stop  = 2.0 * log(0.01);
        double step  = 2.0 * log(0.5);
        // np.arange(start, stop, step) with negative step
        for (double e = start; e > stop && k < MAX_EPS - 1; e += step)
            epsBuf[k++] = (float)exp(e);
        epsBuf[k++] = 1e-4f;  // BLUR**P
        *kBuf = k;
        for (int i = k; i < MAX_EPS; ++i) epsBuf[i] = 1e-4f;
    }
}

// ---------------------------------------------------------------------------
// Fused softmin: out[i] = -eps * log sum_j exp( h_j - C_ij/eps )
//   C_ij = 0.5*sqA_i + 0.5*sqB_j - <A_i, B_j>,  h_j = logw + dual_j/eps
// Dot products via bf16x3-split WMMA with three INDEPENDENT accumulators
// (no D->C chain => no inter-WMMA NOP hazard), summed in VALU afterward.
// Block: 16 rows, 8 waves, each wave owns nB/8 contiguous columns.
// ---------------------------------------------------------------------------
__global__ __launch_bounds__(256)
void softmin_kernel(float* __restrict__ out,
                    const u16* __restrict__ Ahi, const u16* __restrict__ Alo,
                    const float* __restrict__ sqA,
                    const u16* __restrict__ Bhi, const u16* __restrict__ Blo,
                    const float* __restrict__ sqB,
                    const float* __restrict__ dual,   // may be null (init)
                    float logw, int nB,
                    const float* __restrict__ epsBuf,
                    const int* __restrict__ kBuf, int iter) {
    if (iter >= 0 && iter >= *kBuf) return;           // uniform no-op past schedule
    const float eps   = (iter < 0) ? 1e-4f : epsBuf[iter];
    const float LOG2E = 1.4426950408889634f;
    const float LN2   = 0.6931471805599453f;
    const float sEps  = LOG2E / eps;

    __shared__ float cy[4096];
    __shared__ float partM[8][32][8];
    __shared__ float partS[8][32][8];
    __shared__ float rm[16][16];
    __shared__ float rs[16][16];

    const int tid = threadIdx.x;
    // build per-column term cy_j = log2e*logw + sEps*(g_j - 0.5*|y_j|^2)
    for (int j = tid; j < nB; j += 256) {
        float g = dual ? dual[j] : 0.0f;
        cy[j] = fmaf(sEps, g - 0.5f * sqB[j], LOG2E * logw);
    }
    __syncthreads();

    const int wave = tid >> 5, lane = tid & 31;
    const int half = lane >> 4, r = lane & 15;
    const int rowBase = blockIdx.x * 16;

    // A-tile: interleaved 16-bit A layout (half0: K0-7 | K16-23, half1: K8-15 | K24-31)
    const u16* arow_h = Ahi + (size_t)(rowBase + r) * DIMS;
    const u16* arow_l = Alo + (size_t)(rowBase + r) * DIMS;
    V16 aH, aL;
    aH.q[0] = *(const uint4*)(arow_h + half * 8);
    aH.q[1] = *(const uint4*)(arow_h + 16 + half * 8);
    aL.q[0] = *(const uint4*)(arow_l + half * 8);
    aL.q[1] = *(const uint4*)(arow_l + 16 + half * 8);

    const int chunk  = nB >> 3;        // columns per wave
    const int ntiles = chunk >> 4;     // 16-col tiles per wave
    const int colBase0 = wave * chunk;

    float m[8], s[8];
    #pragma unroll
    for (int k = 0; k < 8; ++k) { m[k] = -__builtin_inff(); s[k] = 0.0f; }

    for (int jt = 0; jt < ntiles; ++jt) {
        const int col = colBase0 + jt * 16;
        // B-tile: interleaved 16-bit B layout (half0: K0-7 | K16-23, half1: K8-15 | K24-31)
        const u16* brow_h = Bhi + (size_t)(col + r) * DIMS;
        const u16* brow_l = Blo + (size_t)(col + r) * DIMS;
        V16 bH, bL;
        bH.q[0] = *(const uint4*)(brow_h + half * 8);
        bH.q[1] = *(const uint4*)(brow_h + 16 + half * 8);
        bL.q[0] = *(const uint4*)(brow_l + half * 8);
        bL.q[1] = *(const uint4*)(brow_l + 16 + half * 8);

        v8f z = {0.f, 0.f, 0.f, 0.f, 0.f, 0.f, 0.f, 0.f};
        // independent accumulators: the 3 WMMAs have no mutual RAW hazard
        v8f aHH = __builtin_amdgcn_wmma_f32_16x16x32_bf16(false, aH.v, false, bH.v,
                                                          (short)0, z, false, false);
        v8f aHL = __builtin_amdgcn_wmma_f32_16x16x32_bf16(false, aH.v, false, bL.v,
                                                          (short)0, z, false, false);
        v8f aLH = __builtin_amdgcn_wmma_f32_16x16x32_bf16(false, aL.v, false, bH.v,
                                                          (short)0, z, false, false);

        const float cyv = cy[col + (lane & 15)];
        #pragma unroll
        for (int k = 0; k < 8; ++k) {
            float dot = (aHH[k] + aHL[k]) + aLH[k];
            float t   = fmaf(dot, sEps, cyv);
            float mn  = fmaxf(m[k], t);
            s[k] = s[k] * __builtin_amdgcn_exp2f(m[k] - mn)
                        + __builtin_amdgcn_exp2f(t - mn);
            m[k] = mn;
        }
    }

    #pragma unroll
    for (int k = 0; k < 8; ++k) { partM[wave][lane][k] = m[k]; partS[wave][lane][k] = s[k]; }
    __syncthreads();

    // stage 1: 256 threads; each merges the 8 wave-partials of one (row, lane)
    {
        const int rrow = tid >> 4;           // 0..15 row within tile
        const int idx  = tid & 15;           // 0..15 lane index within half
        const int slot = rrow & 7;
        const int hh   = rrow >> 3;
        const int l    = hh * 16 + idx;
        float M = -__builtin_inff(), S = 0.0f;
        #pragma unroll
        for (int w = 0; w < 8; ++w) {
            float mm = partM[w][l][slot], ss = partS[w][l][slot];
            float mn = fmaxf(M, mm);
            S = S * __builtin_amdgcn_exp2f(M - mn)
              + ss * __builtin_amdgcn_exp2f(mm - mn);
            M = mn;
        }
        rm[rrow][idx] = M;
        rs[rrow][idx] = S;
    }
    __syncthreads();

    // stage 2: 16 threads; merge the 16 lane-partials of each row
    if (tid < 16) {
        float M = -__builtin_inff(), S = 0.0f;
        #pragma unroll
        for (int idx = 0; idx < 16; ++idx) {
            float mm = rm[tid][idx], ss = rs[tid][idx];
            float mn = fmaxf(M, mm);
            S = S * __builtin_amdgcn_exp2f(M - mn)
              + ss * __builtin_amdgcn_exp2f(mm - mn);
            M = mn;
        }
        const int row = rowBase + tid;
        float cx   = -0.5f * sEps * sqA[row];
        float lse2 = cx + M + __builtin_amdgcn_logf(S);
        out[row]   = -eps * LN2 * lse2;
    }
}

// ---------------------------------------------------------------------------
// Damped symmetric update: v = 0.5*(v + v_new) for the 4 potentials
// ---------------------------------------------------------------------------
__global__ __launch_bounds__(256)
void combine_kernel(float* __restrict__ fba, const float* __restrict__ tfba,
                    float* __restrict__ gab, const float* __restrict__ tgab,
                    float* __restrict__ faa, const float* __restrict__ tfaa,
                    float* __restrict__ gbb, const float* __restrict__ tgbb,
                    int n, const int* __restrict__ kBuf, int iter) {
    if (iter >= *kBuf) return;
    int i = blockIdx.x * blockDim.x + threadIdx.x;
    if (i < n) {
        fba[i] = 0.5f * (fba[i] + tfba[i]);
        gab[i] = 0.5f * (gab[i] + tgab[i]);
        faa[i] = 0.5f * (faa[i] + tfaa[i]);
        gbb[i] = 0.5f * (gbb[i] + tgbb[i]);
    }
}

// ---------------------------------------------------------------------------
// result = mean(f_ba_n - f_aa_n) + mean(g_ab_n - g_bb_n)
// ---------------------------------------------------------------------------
__global__ __launch_bounds__(256)
void finalize_kernel(const float* __restrict__ tfba, const float* __restrict__ tgab,
                     const float* __restrict__ tfaa, const float* __restrict__ tgbb,
                     int n, float* __restrict__ out) {
    __shared__ float red[256];
    float acc = 0.0f;
    for (int i = threadIdx.x; i < n; i += 256)
        acc += (tfba[i] - tfaa[i]) + (tgab[i] - tgbb[i]);
    red[threadIdx.x] = acc;
    __syncthreads();
    for (int st = 128; st > 0; st >>= 1) {
        if (threadIdx.x < st) red[threadIdx.x] += red[threadIdx.x + st];
        __syncthreads();
    }
    if (threadIdx.x == 0) out[0] = red[0] / (float)n;
}

// ---------------------------------------------------------------------------
extern "C" void kernel_launch(void* const* d_in, const int* in_sizes, int n_in,
                              void* d_out, int out_size, void* d_ws, size_t ws_size,
                              hipStream_t stream) {
    const float* X = (const float*)d_in[0];   // phi1
    const float* Y = (const float*)d_in[1];   // phi0
    const int n = in_sizes[0] / DIMS;         // 4096

    // workspace carve-up
    char* ws = (char*)d_ws;
    auto take = [&](size_t bytes) { char* p = ws; ws += (bytes + 255) & ~(size_t)255; return p; };
    u16*   Xhi   = (u16*)  take((size_t)n * DIMS * 2);
    u16*   Xlo   = (u16*)  take((size_t)n * DIMS * 2);
    u16*   Yhi   = (u16*)  take((size_t)n * DIMS * 2);
    u16*   Ylo   = (u16*)  take((size_t)n * DIMS * 2);
    float* x2    = (float*)take((size_t)n * 4);
    float* y2    = (float*)take((size_t)n * 4);
    float* g_ab  = (float*)take((size_t)n * 4);
    float* f_ba  = (float*)take((size_t)n * 4);
    float* f_aa  = (float*)take((size_t)n * 4);
    float* g_bb  = (float*)take((size_t)n * 4);
    float* t_fba = (float*)take((size_t)n * 4);
    float* t_gab = (float*)take((size_t)n * 4);
    float* t_faa = (float*)take((size_t)n * 4);
    float* t_gbb = (float*)take((size_t)n * 4);
    float* epsB  = (float*)take(MAX_EPS * 4);
    int*   kBuf  = (int*)  take(4);

    const float logw = -logf((float)n);   // uniform log-weights, N == M
    const dim3 blk(256);
    const dim3 smGrid(n / 16);

    prep_kernel<<<dim3((2 * n + 255) / 256), blk, 0, stream>>>(X, Y, n, Xhi, Xlo, Yhi, Ylo, x2, y2);
    schedule_kernel<<<dim3(1), dim3(1024), 0, stream>>>(X, Y, n, epsB, kBuf);

    // init at eps_list[0]
    softmin_kernel<<<smGrid, blk, 0, stream>>>(g_ab, Yhi, Ylo, y2, Xhi, Xlo, x2, nullptr, logw, n, epsB, kBuf, 0);
    softmin_kernel<<<smGrid, blk, 0, stream>>>(f_ba, Xhi, Xlo, x2, Yhi, Ylo, y2, nullptr, logw, n, epsB, kBuf, 0);
    softmin_kernel<<<smGrid, blk, 0, stream>>>(f_aa, Xhi, Xlo, x2, Xhi, Xlo, x2, nullptr, logw, n, epsB, kBuf, 0);
    softmin_kernel<<<smGrid, blk, 0, stream>>>(g_bb, Yhi, Ylo, y2, Yhi, Ylo, y2, nullptr, logw, n, epsB, kBuf, 0);

    // eps-scaling loop (device-side no-op past the real schedule length)
    for (int it = 0; it < MAX_EPS; ++it) {
        softmin_kernel<<<smGrid, blk, 0, stream>>>(t_fba, Xhi, Xlo, x2, Yhi, Ylo, y2, g_ab, logw, n, epsB, kBuf, it);
        softmin_kernel<<<smGrid, blk, 0, stream>>>(t_gab, Yhi, Ylo, y2, Xhi, Xlo, x2, f_ba, logw, n, epsB, kBuf, it);
        softmin_kernel<<<smGrid, blk, 0, stream>>>(t_faa, Xhi, Xlo, x2, Xhi, Xlo, x2, f_aa, logw, n, epsB, kBuf, it);
        softmin_kernel<<<smGrid, blk, 0, stream>>>(t_gbb, Yhi, Ylo, y2, Yhi, Ylo, y2, g_bb, logw, n, epsB, kBuf, it);
        combine_kernel<<<dim3((n + 255) / 256), blk, 0, stream>>>(f_ba, t_fba, g_ab, t_gab, f_aa, t_faa, g_bb, t_gbb, n, kBuf, it);
    }

    // final extrapolation at eps = blur^2 = 1e-4 (iter = -1 => always run)
    softmin_kernel<<<smGrid, blk, 0, stream>>>(t_fba, Xhi, Xlo, x2, Yhi, Ylo, y2, g_ab, logw, n, epsB, kBuf, -1);
    softmin_kernel<<<smGrid, blk, 0, stream>>>(t_gab, Yhi, Ylo, y2, Xhi, Xlo, x2, f_ba, logw, n, epsB, kBuf, -1);
    softmin_kernel<<<smGrid, blk, 0, stream>>>(t_faa, Xhi, Xlo, x2, Xhi, Xlo, x2, f_aa, logw, n, epsB, kBuf, -1);
    softmin_kernel<<<smGrid, blk, 0, stream>>>(t_gbb, Yhi, Ylo, y2, Yhi, Ylo, y2, g_bb, logw, n, epsB, kBuf, -1);

    finalize_kernel<<<dim3(1), blk, 0, stream>>>(t_fba, t_gab, t_faa, t_gbb, n, (float*)d_out);
}
